// Trump_Encodeur_83305185673473
// MI455X (gfx1250) — matrix operand order repaired and checked by reading, not compiled, not run on hardware
//
#include <hip/hip_runtime.h>
#include <hip/hip_bf16.h>
#include <math.h>

// ---------------------------------------------------------------------------
// Problem constants (match reference)
// ---------------------------------------------------------------------------
#define D_B   128
#define D_T   512
#define D_IN  256
#define D_H   1024
#define D_OUT 1024

typedef __attribute__((ext_vector_type(8)))  __bf16 v8bf;
typedef __attribute__((ext_vector_type(16))) __bf16 v16bf;
typedef __attribute__((ext_vector_type(8)))  float  v8f;

#define WMMA_BF16(A, Bm, C) \
  __builtin_amdgcn_wmma_f32_16x16x32_bf16(false, (A), false, (Bm), (short)0, (C), false, false)

// ---------------------------------------------------------------------------
// Helpers
// ---------------------------------------------------------------------------
__device__ __forceinline__ unsigned short f2bf(float x) {
  union { float f; unsigned u; } v; v.f = x;
  unsigned r = v.u + 0x7FFFu + ((v.u >> 16) & 1u);
  return (unsigned short)(r >> 16);
}

// Build a 16-element bf16 fragment from two 16-byte chunks (ISA VGPR layout:
// elements 0..7 from `lo`, elements 8..15 from `hi`).
__device__ __forceinline__ v16bf frag16(const unsigned short* lo, const unsigned short* hi) {
  v8bf a = *(const v8bf*)lo;
  v8bf b = *(const v8bf*)hi;
  return __builtin_shufflevector(a, b, 0,1,2,3,4,5,6,7,8,9,10,11,12,13,14,15);
}

// ===========================================================================
// Big-GEMM inner loop: one 16-row A strip x 8 16-col LDS weight tiles.
//  - unroll 2: rotates the two A register buffers (kills the a_cur=a_nxt
//    v_mov copies and their WMMA->VALU WAR hazard v_nops). Full unroll
//    previously caused scratch spills; 2 keeps pressure bounded.
//  - Per K-chunk: two groups of 4 B fragments; <=32 B VGPRs live at once,
//    group-1 ds_loads overlap group-0 WMMAs.
// ===========================================================================
template <int K, int LDSTR>
__device__ __forceinline__ void gemm_tile8(
    const unsigned short* __restrict__ ap,     // A row base (lane mapping applied)
    const unsigned short* __restrict__ bbase,  // LDS &bs[lr*LDSTR + lh*16]
    v8f* __restrict__ acc)
{
  v16bf a_cur = frag16(ap, ap + 8);
#pragma unroll 2
  for (int k0 = 0; k0 < K; k0 += 32) {
    const int kn = (k0 + 32) & (K - 1);        // wraps on last iter (benign read)
    v16bf bfr[4];
#pragma unroll
    for (int j = 0; j < 4; ++j) {
      const unsigned short* bp = bbase + (unsigned)(16 * j) * LDSTR + k0;
      bfr[j] = frag16(bp, bp + 8);
    }
    v16bf a_nxt = frag16(ap + kn, ap + kn + 8);
#pragma unroll
    for (int j = 0; j < 4; ++j) acc[j] = WMMA_BF16(a_cur, bfr[j], acc[j]);
#pragma unroll
    for (int j = 0; j < 4; ++j) {
      const unsigned short* bp = bbase + (unsigned)(16 * (j + 4)) * LDSTR + k0;
      bfr[j] = frag16(bp, bp + 8);
    }
#pragma unroll
    for (int j = 0; j < 4; ++j) acc[j + 4] = WMMA_BF16(a_cur, bfr[j], acc[j + 4]);
    a_cur = a_nxt;
  }
}

// Stage NROWS x K bf16 weight rows (output cols n0..) into LDS.
template <int K, int LDSTR, int NROWS>
__device__ __forceinline__ void stage_weights(
    const unsigned short* __restrict__ W, unsigned short* __restrict__ ws_lds,
    int n0, int tid) {
  for (int idx = tid * 8; idx < NROWS * K; idx += 256 * 8) {
    int r = idx / K, k = idx % K;
    *(v8bf*)&ws_lds[(unsigned)r * LDSTR + k] = *(const v8bf*)&W[(size_t)(n0 + r) * K + k];
  }
}

// ---------------------------------------------------------------------------
// fp32 -> bf16 conversion
// ---------------------------------------------------------------------------
__global__ void cvt_bf16(const float* __restrict__ in, unsigned short* __restrict__ out, int n) {
  int i = blockIdx.x * blockDim.x + threadIdx.x;
  if (i < n) out[i] = f2bf(in[i]);
}

// Zero H ping-pong buffers, the grid-sync counter, and out[:, 0, :].
__global__ void init_state(unsigned short* __restrict__ h, float* __restrict__ out,
                           unsigned* __restrict__ cnt) {
  int i = blockIdx.x * blockDim.x + threadIdx.x;
  if (i < 2 * D_B * D_H) h[i] = 0;
  if (i < D_B * D_OUT) {
    int b = i >> 10, n = i & (D_OUT - 1);
    out[(size_t)b * ((D_T + 1) * D_OUT) + n] = 0.0f;
  }
  if (i == 0) *cnt = 0u;
}

// ---------------------------------------------------------------------------
// GEMM 1: Xproj[t*B+b][h] = seq[b][t][:] . Wx[h][:] + Wx_b[h]  (M=65536,K=256)
// Block: 128 output cols x 512 output rows (4 M-subtiles, wave -> 16x128).
// ---------------------------------------------------------------------------
#define XP_LDSTR (D_IN + 8)   // 264

__global__ __launch_bounds__(256) void xproj_gemm(
    const unsigned short* __restrict__ seqb, const unsigned short* __restrict__ Wxb,
    const float* __restrict__ Wx_b, float* __restrict__ Xp) {
  __shared__ unsigned short bs[128 * XP_LDSTR];    // 67.6 KB
  const int tid  = threadIdx.x;
  const int lane = tid & 31, wave = tid >> 5;
  const int lr = lane & 15, lh = lane >> 4;
  const int n0 = blockIdx.x * 128;

  stage_weights<D_IN, XP_LDSTR, 128>(Wxb, bs, n0, tid);
  __syncthreads();

  const unsigned short* bbase = &bs[(unsigned)lr * XP_LDSTR + lh * 16];

#pragma unroll 1
  for (int msub = 0; msub < 4; ++msub) {
    const int m0 = blockIdx.y * 512 + msub * 128 + wave * 16;
    // A row r = t*B + b maps to seq element (b, t, :)
    const int arow = m0 + lr;
    const unsigned short* ap =
        seqb + ((size_t)(arow & (D_B - 1)) * D_T + (arow >> 7)) * D_IN + lh * 16;
    v8f acc[8] = {};
    gemm_tile8<D_IN, XP_LDSTR>(ap, bbase, acc);
#pragma unroll
    for (int j = 0; j < 8; ++j)
#pragma unroll
      for (int i = 0; i < 8; ++i) {
        int row = m0 + lh * 8 + i;
        int n   = n0 + 16 * j + lr;
        Xp[(size_t)row * D_H + n] = acc[j][i] + Wx_b[n];
      }
  }
}

// ---------------------------------------------------------------------------
// Persistent recurrent scan: 32 blocks, each owns a 32-wide column slice.
// ---------------------------------------------------------------------------
#define SCAN_BLOCKS   (D_H / 32)   // 32
#define WH_LDS_STRIDE 1032         // 1024 + 8 bf16 pad -> conflict-free ds_load_b128

__global__ __launch_bounds__(256) void scan_kernel(
    const unsigned short* __restrict__ Whb, const float* __restrict__ Wh_bias,
    const float* __restrict__ Xp, unsigned short* __restrict__ Sb,
    unsigned short* __restrict__ H, unsigned* __restrict__ cnt) {
  __shared__ unsigned short whs[32 * WH_LDS_STRIDE];   // 64.5 KB

  const int tid  = threadIdx.x;
  const int lane = tid & 31, wave = tid >> 5;
  const int lr = lane & 15, lh = lane >> 4;
  const int n0 = blockIdx.x * 32;
  const int m0 = wave * 16;

  // Stage this block's 32x1024 Wh slice into LDS once (reused for 512 steps).
  for (int idx = tid * 8; idx < 32 * D_H; idx += 256 * 8) {
    int r = idx >> 10, k = idx & (D_H - 1);
    *(v8bf*)&whs[r * WH_LDS_STRIDE + k] = *(const v8bf*)&Whb[(size_t)(n0 + r) * D_H + k];
  }
  __syncthreads();

  const unsigned short* bbase = &whs[(unsigned)lr * WH_LDS_STRIDE + lh * 16];
  unsigned short* Hbuf0 = H;
  unsigned short* Hbuf1 = H + (size_t)D_B * D_H;

  for (int t = 0; t < D_T; ++t) {
    const unsigned short* Hin  = (t & 1) ? Hbuf1 : Hbuf0;
    unsigned short*       Hout = (t & 1) ? Hbuf0 : Hbuf1;

    // --- pipelined 128x32 = (128x1024) x (1024x32) step-GEMM ---
    const unsigned short* ap = Hin + (size_t)(m0 + lr) * D_H + lh * 16;
    v8f acc[2] = {};
    {
      v16bf a_cur = frag16(ap, ap + 8);
#pragma unroll 4
      for (int k0 = 0; k0 < D_H; k0 += 32) {
        const int kn = (k0 + 32) & (D_H - 1);
        const unsigned short* b0 = bbase + k0;
        const unsigned short* b1 = bbase + (unsigned)16 * WH_LDS_STRIDE + k0;
        v16bf f0 = frag16(b0, b0 + 8);
        v16bf f1 = frag16(b1, b1 + 8);
        v16bf a_nxt = frag16(ap + kn, ap + kn + 8);
        acc[0] = WMMA_BF16(a_cur, f0, acc[0]);
        acc[1] = WMMA_BF16(a_cur, f1, acc[1]);
        a_cur = a_nxt;
      }
    }

#pragma unroll
    for (int j = 0; j < 2; ++j)
#pragma unroll
      for (int i = 0; i < 8; ++i) {
        int row = m0 + lh * 8 + i;               // batch index
        int n   = n0 + 16 * j + lr;
        size_t o = ((size_t)t * D_B + row) * D_H + n;
        float s = acc[j][i] + Wh_bias[n] + Xp[o];
        Sb[o] = f2bf(s);                          // pre-activation for Wo GEMM
        Hout[(size_t)row * D_H + n] = f2bf(tanhf(s));
        // Pull next timestep's x-projection toward the WGP (global_prefetch_b8)
        if (t + 1 < D_T) __builtin_prefetch(&Xp[o + (size_t)D_B * D_H], 0, 0);
      }

    // ---- device-wide barrier (monotonic counter, no reset race) ----
    __threadfence();
    __syncthreads();
    if (tid == 0) {
      atomicAdd(cnt, 1u);
      unsigned target = (unsigned)gridDim.x * (unsigned)(t + 1);
      while (atomicAdd(cnt, 0u) < target) __builtin_amdgcn_s_sleep(2);
    }
    __syncthreads();
    __threadfence();
  }
}

// ---------------------------------------------------------------------------
// GEMM 3: out[b][t+1][n] = S[t*B+b][:] . Wo[n][:] + Wo_b[n]  (M=65536,K=1024)
// Block: 128 output cols (halves S re-read traffic vs 64) x 512 rows.
// ---------------------------------------------------------------------------
#define OG_LDSTR (D_H + 8)    // 1032

__global__ __launch_bounds__(256) void out_gemm(
    const unsigned short* __restrict__ Sb, const unsigned short* __restrict__ Wob,
    const float* __restrict__ Wo_b, float* __restrict__ out) {
  __shared__ unsigned short bs[128 * OG_LDSTR];   // 264 KB (WGP has 320 KB)
  const int tid  = threadIdx.x;
  const int lane = tid & 31, wave = tid >> 5;
  const int lr = lane & 15, lh = lane >> 4;
  const int n0 = blockIdx.x * 128;

  stage_weights<D_H, OG_LDSTR, 128>(Wob, bs, n0, tid);
  __syncthreads();

  const unsigned short* bbase = &bs[(unsigned)lr * OG_LDSTR + lh * 16];

#pragma unroll 1
  for (int msub = 0; msub < 4; ++msub) {
    const int m0 = blockIdx.y * 512 + msub * 128 + wave * 16;
    const unsigned short* ap = Sb + (size_t)(m0 + lr) * D_H + lh * 16;
    v8f acc[8] = {};
    gemm_tile8<D_H, OG_LDSTR>(ap, bbase, acc);
#pragma unroll
    for (int j = 0; j < 8; ++j)
#pragma unroll
      for (int i = 0; i < 8; ++i) {
        int row = m0 + lh * 8 + i;                // r = t*B + b
        int b   = row & (D_B - 1);
        int t   = row >> 7;
        int n   = n0 + 16 * j + lr;
        out[(size_t)b * ((D_T + 1) * D_OUT) + (size_t)(t + 1) * D_OUT + n] =
            acc[j][i] + Wo_b[n];
      }
  }
}

// ---------------------------------------------------------------------------
// Launch
// ---------------------------------------------------------------------------
extern "C" void kernel_launch(void* const* d_in, const int* in_sizes, int n_in,
                              void* d_out, int out_size, void* d_ws, size_t ws_size,
                              hipStream_t stream) {
  (void)in_sizes; (void)n_in; (void)out_size; (void)ws_size;

  const float* seq  = (const float*)d_in[0];
  const float* Wh_w = (const float*)d_in[1];
  const float* Wh_b = (const float*)d_in[2];
  const float* Wx_w = (const float*)d_in[3];
  const float* Wx_b = (const float*)d_in[4];
  const float* Wo_w = (const float*)d_in[5];
  const float* Wo_b = (const float*)d_in[6];
  float* out = (float*)d_out;

  char* ws = (char*)d_ws;
  size_t off = 0;
  float*          Xp   = (float*)(ws + off);          off += (size_t)D_T * D_B * D_H * 4;  // 256 MB
  unsigned short* Sb   = (unsigned short*)(ws + off); off += (size_t)D_T * D_B * D_H * 2;  // 128 MB
  unsigned short* seqb = (unsigned short*)(ws + off); off += (size_t)D_B * D_T * D_IN * 2; //  32 MB
  unsigned short* Wxb  = (unsigned short*)(ws + off); off += (size_t)D_H * D_IN * 2;
  unsigned short* Whb  = (unsigned short*)(ws + off); off += (size_t)D_H * D_H * 2;
  unsigned short* Wob  = (unsigned short*)(ws + off); off += (size_t)D_OUT * D_H * 2;
  unsigned short* Hb   = (unsigned short*)(ws + off); off += (size_t)2 * D_B * D_H * 2;    // ping-pong H
  unsigned*       cnt  = (unsigned*)(ws + off);       off += 256;

  const int nseq = D_B * D_T * D_IN;
  cvt_bf16<<<(nseq + 255) / 256, 256, 0, stream>>>(seq, seqb, nseq);
  cvt_bf16<<<(D_H * D_IN + 255) / 256, 256, 0, stream>>>(Wx_w, Wxb, D_H * D_IN);
  cvt_bf16<<<(D_H * D_H + 255) / 256, 256, 0, stream>>>(Wh_w, Whb, D_H * D_H);
  cvt_bf16<<<(D_OUT * D_H + 255) / 256, 256, 0, stream>>>(Wo_w, Wob, D_OUT * D_H);

  init_state<<<(2 * D_B * D_H + 255) / 256, 256, 0, stream>>>(Hb, out, cnt);

  // Input projection for all timesteps (parallel GEMM).
  xproj_gemm<<<dim3(D_H / 128, (D_T * D_B) / 512), 256, 0, stream>>>(seqb, Wxb, Wx_b, Xp);

  // Persistent recurrent scan with device-wide barrier per timestep.
  scan_kernel<<<SCAN_BLOCKS, 256, 0, stream>>>(Whb, Wh_b, Xp, Sb, Hb, cnt);

  // Deferred output projection on all pre-activations (parallel GEMM).
  out_gemm<<<dim3(D_OUT / 128, (D_T * D_B) / 512), 256, 0, stream>>>(Sb, Wob, Wo_b, out);
}
